// GroupQueryAttention_88845693485860
// MI455X (gfx1250) — matrix operand order repaired and checked by reading, not compiled, not run on hardware
//
#include <hip/hip_runtime.h>
#include <hip/hip_bf16.h>
#include <math.h>
#include <stdint.h>

typedef __attribute__((ext_vector_type(16))) _Float16 v16h;
typedef __attribute__((ext_vector_type(8)))  _Float16 v8h;
typedef __attribute__((ext_vector_type(4)))  _Float16 v4h;
typedef __attribute__((ext_vector_type(8)))  float    v8f;

#define B_   2
#define S_   2048
#define E_   2048
#define H_   16
#define KVH_ 4
#define HD_  128
#define REP_ 4

static __device__ __forceinline__ v8f wmma16(v16h a, v16h b, v8f c) {
  // D = A(16x32 f16) * B(32x16 f16) + C(16x16 f32)
  return __builtin_amdgcn_wmma_f32_16x16x32_f16(false, a, false, b, (short)0, c,
                                                false, false);
}

// gfx1250 async copy: 16B per lane, global -> LDS, tracked by ASYNCcnt.
// GVS mode: lds-dest VGPR, 32-bit byte-offset VGPR, 64-bit SGPR base.
static __device__ __forceinline__ void async_b128(unsigned lds, unsigned voff,
                                                  const void* base) {
  asm volatile("global_load_async_to_lds_b128 %0, %1, %2"
               :: "v"(lds), "v"(voff), "s"(base) : "memory");
}
static __device__ __forceinline__ unsigned lds_addr(const void* p) {
  // low 32 bits of a generic pointer into LDS == LDS byte address
  return (unsigned)(uintptr_t)p;
}

// ---------------------------------------------------------------------------
// GEMM: out[M,N] = A[M,K] * Wt[N,K]^T + bias[N]; A,Wt f16; out f32 or f16.
// Block 64x128, 8 waves (2x4) each 32x32; BK=32; double-buffered async stage.
// ---------------------------------------------------------------------------
template <bool OUT_F32>
__global__ __launch_bounds__(256)
void gemm_f16(const _Float16* __restrict__ A,   // [M][K]
              const _Float16* __restrict__ Wt,  // [N][K]
              const float* __restrict__ bias,
              void* __restrict__ outp,
              int M, int N, int K) {
  __shared__ __align__(16) _Float16 sA[2][64][32];
  __shared__ __align__(16) _Float16 sB[2][128][32];

  const int tid  = threadIdx.x;
  const int lane = tid & 31;
  const int wid  = tid >> 5;
  const int kh   = lane >> 4;
  const int ln   = lane & 15;
  const int waveM = wid & 1;
  const int waveN = wid >> 1;
  const int m0 = blockIdx.x * 64;
  const int n0 = blockIdx.y * 128;

  // 16B-chunk copy assignments (1 A-chunk + 2 B-chunks per thread per tile)
  const unsigned aL[2] = { lds_addr(&sA[0][0][0]) + (unsigned)tid * 16u,
                           lds_addr(&sA[1][0][0]) + (unsigned)tid * 16u };
  const unsigned bL[2] = { lds_addr(&sB[0][0][0]) + (unsigned)tid * 16u,
                           lds_addr(&sB[1][0][0]) + (unsigned)tid * 16u };
  const unsigned aVo  = (unsigned)((( tid >> 2)        * K + (tid & 3) * 8) * 2);
  const unsigned bVo0 = aVo;                                   // rows 0..63
  const unsigned bVo1 = (unsigned)(((( tid >> 2) + 64) * K + (tid & 3) * 8) * 2);

  const _Float16* aBase = A  + (size_t)m0 * K;
  const _Float16* bBase = Wt + (size_t)n0 * K;

  auto stage = [&](int buf, int kt) {
    const _Float16* ab = aBase + kt;
    const _Float16* bb = bBase + kt;
    async_b128(aL[buf],          aVo,  ab);
    async_b128(bL[buf],          bVo0, bb);
    async_b128(bL[buf] + 4096u,  bVo1, bb);
  };

  v8f acc[2][2];
  const v8f vzero = {0.f,0.f,0.f,0.f,0.f,0.f,0.f,0.f};
#pragma unroll
  for (int tm = 0; tm < 2; ++tm)
#pragma unroll
    for (int tn = 0; tn < 2; ++tn) acc[tm][tn] = vzero;

  stage(0, 0);
  int cur = 0;
  for (int kt = 0; kt < K; kt += 32) {
    const bool more = (kt + 32) < K;
    if (more) stage(cur ^ 1, kt + 32);
    if (more) asm volatile("s_wait_asynccnt 3" ::: "memory");  // cur done, next in flight
    else      asm volatile("s_wait_asynccnt 0" ::: "memory");
    __syncthreads();

    v16h aF[2], bF[2];
#pragma unroll
    for (int tm = 0; tm < 2; ++tm) {
      const _Float16* p = &sA[cur][waveM * 32 + tm * 16 + ln][0];
#pragma unroll
      for (int j = 0; j < 8; ++j) aF[tm][j]     = p[8 * kh + j];
#pragma unroll
      for (int j = 0; j < 8; ++j) aF[tm][8 + j] = p[16 + 8 * kh + j];
    }
#pragma unroll
    for (int tn = 0; tn < 2; ++tn) {
      const _Float16* p = &sB[cur][waveN * 32 + tn * 16 + ln][0];
#pragma unroll
      for (int j = 0; j < 16; ++j) bF[tn][j] = p[j < 8 ? 8 * kh + j : 16 + 8 * kh + (j - 8)];
    }
#pragma unroll
    for (int tm = 0; tm < 2; ++tm)
#pragma unroll
      for (int tn = 0; tn < 2; ++tn)
        acc[tm][tn] = wmma16(aF[tm], bF[tn], acc[tm][tn]);
    __syncthreads();
    cur ^= 1;
  }

#pragma unroll
  for (int tm = 0; tm < 2; ++tm)
#pragma unroll
    for (int tn = 0; tn < 2; ++tn) {
      int gcol = n0 + waveN * 32 + tn * 16 + ln;
      float bv = bias[gcol];
#pragma unroll
      for (int i = 0; i < 8; ++i) {
        int grow = m0 + waveM * 32 + tm * 16 + i + 8 * kh;
        float v = acc[tm][tn][i] + bv;
        if (OUT_F32) ((float*)outp)[(size_t)grow * N + gcol] = v;
        else         ((_Float16*)outp)[(size_t)grow * N + gcol] = (_Float16)v;
      }
    }
}

// ---------------------------------------------------------------------------
// One-time converts / permutes (bandwidth passes, done once, not per tile).
// ---------------------------------------------------------------------------
__global__ void convert_f16v(const float* __restrict__ in,
                             _Float16* __restrict__ out, long n4) {
  long i = (long)blockIdx.x * 256 + threadIdx.x;
  if (i >= n4) return;
  float4 v = ((const float4*)in)[i];
  v4h r; r[0]=(_Float16)v.x; r[1]=(_Float16)v.y; r[2]=(_Float16)v.z; r[3]=(_Float16)v.w;
  ((v4h*)out)[i] = r;
}

// W f32 [K][N] -> Wt f16 [N][K]
__global__ __launch_bounds__(256)
void transpose_convert(const float* __restrict__ W, _Float16* __restrict__ Wt,
                       int K, int N) {
  __shared__ float tile[32][33];
  const int k0 = blockIdx.x * 32, n0 = blockIdx.y * 32;
  const int tx = threadIdx.x & 31, ty = threadIdx.x >> 5;
#pragma unroll
  for (int i = ty; i < 32; i += 8)
    tile[i][tx] = W[(size_t)(k0 + i) * N + n0 + tx];
  __syncthreads();
#pragma unroll
  for (int i = ty; i < 32; i += 8)
    Wt[(size_t)(n0 + i) * K + k0 + tx] = (_Float16)tile[tx][i];
}

// RoPE in-place on f16 Q (b,s,H,HD); reference freqs theta^(-2p)
__global__ void rope_q(_Float16* __restrict__ t, long totalPairs) {
  long idx = (long)blockIdx.x * 256 + threadIdx.x;
  if (idx >= totalPairs) return;
  int p = (int)(idx & 63);
  int s = (int)((idx >> 6) / H_ % S_);
  float freq = __expf(-(float)(2 * p) * __logf(100000.0f));
  float c, sn; __sincosf((float)s * freq, &sn, &c);
  _Float16* b = t + 2 * idx;
  float tr = (float)b[0], ti = (float)b[1];
  b[0] = (_Float16)(tr * c - ti * sn);
  b[1] = (_Float16)(tr * sn + ti * c);
}

// RoPE + permute K: (b,s,g,d) f16 -> (b,g,s,d) f16
__global__ void rope_k_permute(const _Float16* __restrict__ kn,
                               _Float16* __restrict__ kp, long totalPairs) {
  long idx = (long)blockIdx.x * 256 + threadIdx.x;
  if (idx >= totalPairs) return;
  int p = (int)(idx & 63);
  long r = idx >> 6;                 // (b*S+s)*KVH + g
  int g = (int)(r % KVH_);
  long bs = r / KVH_;
  int s = (int)(bs % S_);
  int b = (int)(bs / S_);
  float freq = __expf(-(float)(2 * p) * __logf(100000.0f));
  float c, sn; __sincosf((float)s * freq, &sn, &c);
  const _Float16* src = kn + 2 * idx;
  float tr = (float)src[0], ti = (float)src[1];
  size_t o = ((size_t)(b * KVH_ + g) * S_ + s) * HD_ + 2 * p;
  kp[o]     = (_Float16)(tr * c - ti * sn);
  kp[o + 1] = (_Float16)(tr * sn + ti * c);
}

// V: (b,s,g,d) f16 -> (b,g,d,s) f16
__global__ void permute_v(const _Float16* __restrict__ vn,
                          _Float16* __restrict__ vt, long total) {
  long idx = (long)blockIdx.x * 256 + threadIdx.x;
  if (idx >= total) return;
  int hd = (int)(idx % HD_);
  long r = idx / HD_;
  int g = (int)(r % KVH_);
  long bs = r / KVH_;
  int s = (int)(bs % S_);
  int b = (int)(bs / S_);
  vt[((size_t)(b * KVH_ + g) * HD_ + hd) * S_ + s] = vn[idx];
}

// ---------------------------------------------------------------------------
// Flash GQA attention: block = 32 queries x 1 kv-head x 1 batch; 8 waves =
// 4 q-heads x 2 q-subtiles sharing double-buffered async-staged K/V tiles.
// Q (b,s,H,d) f16; Kp (b,g,s,d) f16; Vt (b,g,d,s) f16; O (b,s,H,d) f16.
// ---------------------------------------------------------------------------
__global__ __launch_bounds__(256)
void gqa_flash(const _Float16* __restrict__ Q, const _Float16* __restrict__ Kp,
               const _Float16* __restrict__ Vt, _Float16* __restrict__ O) {
  __shared__ __align__(16) _Float16 sK[2][32][128];   // [key][hd]
  __shared__ __align__(16) _Float16 sV[2][128][32];   // [hd][key]
  __shared__ __align__(16) _Float16 sP[8][16][32];    // per-wave P staging

  const int tid  = threadIdx.x;
  const int lane = tid & 31;
  const int wid  = tid >> 5;
  const int kh   = lane >> 4;
  const int ln   = lane & 15;
  const int qb = blockIdx.x;
  const int g  = blockIdx.y;
  const int bb = blockIdx.z;
  const int h  = g * REP_ + (wid >> 1);
  const int qbase = qb * 32 + (wid & 1) * 16;
  const int qrow  = qbase + ln;

  // Q fragments: direct vector loads from f16 global (A layout)
  v16h aQ[4];
  const _Float16* qp = Q + ((size_t)(bb * S_ + qrow) * H_ + h) * HD_;
#pragma unroll
  for (int c = 0; c < 4; ++c) {
    v8h q0 = *(const v8h*)(qp + 32 * c + 8 * kh);
    v8h q1 = *(const v8h*)(qp + 32 * c + 16 + 8 * kh);
#pragma unroll
    for (int j = 0; j < 8; ++j) { aQ[c][j] = q0[j]; aQ[c][8 + j] = q1[j]; }
  }

  // async staging setup: K tile contiguous 8KB; V tile rows of 64B
  const _Float16* kPlane = Kp + (size_t)(bb * KVH_ + g) * S_ * HD_;
  const _Float16* vPlane = Vt + (size_t)(bb * KVH_ + g) * HD_ * S_;
  const unsigned kL[2] = { lds_addr(&sK[0][0][0]) + (unsigned)tid * 16u,
                           lds_addr(&sK[1][0][0]) + (unsigned)tid * 16u };
  const unsigned vL[2] = { lds_addr(&sV[0][0][0]) + (unsigned)tid * 16u,
                           lds_addr(&sV[1][0][0]) + (unsigned)tid * 16u };
  const unsigned kVo0 = (unsigned)tid * 16u;                 // contiguous
  const unsigned kVo1 = kVo0 + 4096u;
  const unsigned vVo0 = (unsigned)(((tid >> 2) * S_ + (tid & 3) * 8) * 2);
  const unsigned vVo1 = vVo0 + (unsigned)(64 * S_ * 2);

  auto stage = [&](int buf, int kt) {
    const _Float16* kB = kPlane + (size_t)kt * 32 * HD_;  // contiguous tile
    const _Float16* vB = vPlane + kt * 32;                // column offset
    async_b128(kL[buf],          kVo0, kB);
    async_b128(kL[buf] + 4096u,  kVo1, kB);
    async_b128(vL[buf],          vVo0, vB);
    async_b128(vL[buf] + 4096u,  vVo1, vB);
  };

  const float NEG_INF = -__builtin_inff();
  float mrow[8], lrow[8];
  v8f acc[8];
  const v8f vzero = {0.f,0.f,0.f,0.f,0.f,0.f,0.f,0.f};
#pragma unroll
  for (int i = 0; i < 8; ++i) { mrow[i] = NEG_INF; lrow[i] = 0.f; }
#pragma unroll
  for (int t = 0; t < 8; ++t) acc[t] = vzero;

  const float scale = 0.08838834764831845f;  // 1/sqrt(128)

  stage(0, 0);
  int cur = 0;
  for (int kt = 0; kt <= qb; ++kt) {
    const int kb = kt * 32;
    const bool more = kt < qb;
    if (more) stage(cur ^ 1, kt + 1);
    if (more) asm volatile("s_wait_asynccnt 4" ::: "memory");
    else      asm volatile("s_wait_asynccnt 0" ::: "memory");
    __syncthreads();

    // S = Q K^T (two 16-key subtiles, K-dim 128 = 4 chunks)
    v8f sc[2];
#pragma unroll
    for (int nt = 0; nt < 2; ++nt) {
      v8f s = vzero;
#pragma unroll
      for (int c = 0; c < 4; ++c) {
        v16h bK;
        const _Float16* p = &sK[cur][nt * 16 + ln][32 * c];
#pragma unroll
        for (int j = 0; j < 16; ++j) bK[j] = p[j < 8 ? 8 * kh + j : 16 + 8 * kh + (j - 8)];
        s = wmma16(aQ[c], bK, s);
      }
      sc[nt] = s;
    }
    // scale + causal mask
#pragma unroll
    for (int nt = 0; nt < 2; ++nt) {
      int kg = kb + nt * 16 + ln;
#pragma unroll
      for (int i = 0; i < 8; ++i) {
        int qg = qbase + i + 8 * kh;
        float v = sc[nt][i] * scale;
        sc[nt][i] = (kg <= qg) ? v : NEG_INF;
      }
    }
    // online softmax (row reductions within 16-lane halves)
#pragma unroll
    for (int i = 0; i < 8; ++i) {
      float mx = fmaxf(sc[0][i], sc[1][i]);
#pragma unroll
      for (int off = 8; off >= 1; off >>= 1)
        mx = fmaxf(mx, __shfl_xor(mx, off, 16));
      float mnew = fmaxf(mrow[i], mx);
      float alpha = __expf(mrow[i] - mnew);
      float p0 = __expf(sc[0][i] - mnew);
      float p1 = __expf(sc[1][i] - mnew);
      sc[0][i] = p0; sc[1][i] = p1;
      float rs = p0 + p1;
#pragma unroll
      for (int off = 8; off >= 1; off >>= 1)
        rs += __shfl_xor(rs, off, 16);
      lrow[i] = lrow[i] * alpha + rs;
      mrow[i] = mnew;
#pragma unroll
      for (int t = 0; t < 8; ++t) acc[t][i] *= alpha;
    }

    // P: D-layout f32 -> A-layout f16 via wave-private LDS
#pragma unroll
    for (int nt = 0; nt < 2; ++nt)
#pragma unroll
      for (int i = 0; i < 8; ++i)
        sP[wid][i + 8 * kh][nt * 16 + ln] = (_Float16)sc[nt][i];
    asm volatile("s_wait_dscnt 0" ::: "memory");

    v16h aP;
    {
      const _Float16* pp = &sP[wid][ln][0];
#pragma unroll
      for (int j = 0; j < 8; ++j) aP[j]     = pp[8 * kh + j];
#pragma unroll
      for (int j = 0; j < 8; ++j) aP[8 + j] = pp[16 + 8 * kh + j];
    }

    // O += P V (8 hd-column tiles)
#pragma unroll
    for (int t = 0; t < 8; ++t) {
      v16h bV;
      const _Float16* p = &sV[cur][t * 16 + ln][0];
#pragma unroll
      for (int j = 0; j < 16; ++j) bV[j] = p[j < 8 ? 8 * kh + j : 16 + 8 * kh + (j - 8)];
      acc[t] = wmma16(aP, bV, acc[t]);
    }
    __syncthreads();
    cur ^= 1;
  }

  // normalize + store f16 (b,s,H,d) -> rows of the final GEMM
  float inv[8];
#pragma unroll
  for (int i = 0; i < 8; ++i) inv[i] = 1.0f / lrow[i];
#pragma unroll
  for (int t = 0; t < 8; ++t)
#pragma unroll
    for (int i = 0; i < 8; ++i) {
      int row = qbase + i + 8 * kh;
      O[((size_t)(bb * S_ + row) * H_ + h) * HD_ + t * 16 + ln] =
          (_Float16)(acc[t][i] * inv[i]);
    }
}

// ---------------------------------------------------------------------------
extern "C" void kernel_launch(void* const* d_in, const int* in_sizes, int n_in,
                              void* d_out, int out_size, void* d_ws, size_t ws_size,
                              hipStream_t stream) {
  const float* x  = (const float*)d_in[0];
  const float* wq = (const float*)d_in[1];
  const float* bq = (const float*)d_in[2];
  const float* wk = (const float*)d_in[3];
  const float* bk = (const float*)d_in[4];
  const float* wv = (const float*)d_in[5];
  const float* bv = (const float*)d_in[6];
  const float* wo = (const float*)d_in[7];
  const float* bo = (const float*)d_in[8];
  float* out = (float*)d_out;

  const int M = B_ * S_;
  _Float16* ws = (_Float16*)d_ws;
  _Float16* xh   = ws;                          ws += (size_t)M * E_;
  _Float16* wqt  = ws;                          ws += (size_t)E_ * (H_ * HD_);
  _Float16* wkt  = ws;                          ws += (size_t)E_ * (KVH_ * HD_);
  _Float16* wvt  = ws;                          ws += (size_t)E_ * (KVH_ * HD_);
  _Float16* wot  = ws;                          ws += (size_t)(H_ * HD_) * E_;
  _Float16* qh   = ws;                          ws += (size_t)M * H_ * HD_;
  _Float16* khn  = ws;                          ws += (size_t)M * KVH_ * HD_;
  _Float16* vhn  = ws;                          ws += (size_t)M * KVH_ * HD_;
  _Float16* khp  = ws;                          ws += (size_t)M * KVH_ * HD_;
  _Float16* vT   = ws;                          ws += (size_t)M * KVH_ * HD_;
  _Float16* abuf = ws;

  dim3 blk(256);

  // one-time conversions
  long xn4 = (long)M * E_ / 4;
  convert_f16v<<<dim3((unsigned)((xn4 + 255) / 256)), blk, 0, stream>>>(x, xh, xn4);
  transpose_convert<<<dim3(E_ / 32, (H_ * HD_) / 32),  blk, 0, stream>>>(wq, wqt, E_, H_ * HD_);
  transpose_convert<<<dim3(E_ / 32, (KVH_ * HD_) / 32), blk, 0, stream>>>(wk, wkt, E_, KVH_ * HD_);
  transpose_convert<<<dim3(E_ / 32, (KVH_ * HD_) / 32), blk, 0, stream>>>(wv, wvt, E_, KVH_ * HD_);
  transpose_convert<<<dim3((H_ * HD_) / 32, E_ / 32),  blk, 0, stream>>>(wo, wot, H_ * HD_, E_);

  // QKV projections (f16 out)
  gemm_f16<false><<<dim3(M / 64, (H_ * HD_) / 128), blk, 0, stream>>>(
      xh, wqt, bq, qh, M, H_ * HD_, E_);
  gemm_f16<false><<<dim3(M / 64, (KVH_ * HD_) / 128), blk, 0, stream>>>(
      xh, wkt, bk, khn, M, KVH_ * HD_, E_);
  gemm_f16<false><<<dim3(M / 64, (KVH_ * HD_) / 128), blk, 0, stream>>>(
      xh, wvt, bv, vhn, M, KVH_ * HD_, E_);

  // RoPE + layout permutes
  long qpairs = (long)M * H_ * (HD_ / 2);
  rope_q<<<dim3((unsigned)((qpairs + 255) / 256)), blk, 0, stream>>>(qh, qpairs);
  long kpairs = (long)M * KVH_ * (HD_ / 2);
  rope_k_permute<<<dim3((unsigned)((kpairs + 255) / 256)), blk, 0, stream>>>(khn, khp, kpairs);
  long vtot = (long)M * KVH_ * HD_;
  permute_v<<<dim3((unsigned)((vtot + 255) / 256)), blk, 0, stream>>>(vhn, vT, vtot);

  // flash attention (f16 out)
  gqa_flash<<<dim3(S_ / 32, KVH_, B_), blk, 0, stream>>>(qh, khp, vT, abuf);

  // output projection (f32 out)
  gemm_f16<true><<<dim3(M / 64, E_ / 128), blk, 0, stream>>>(
      abuf, wot, bo, out, M, E_, E_);
}
